// GlobalAddAttention_69483980915148
// MI455X (gfx1250) — matrix operand order repaired and checked by reading, not compiled, not run on hardware
//
#include <hip/hip_runtime.h>

#define E_DIM 512
#define NHEAD 8
#define NB    64
#define NNODE 255
#define LSEQ  256
#define DH    64
#define LP    (LSEQ + 8)   // padded LDS row (528B stride: 16B-aligned, conflict-free)

typedef __attribute__((ext_vector_type(16))) __bf16 bf16x16;
typedef __attribute__((ext_vector_type(8)))  __bf16 bf16x8;
typedef __attribute__((ext_vector_type(8)))  float  f32x8;

static __device__ inline bf16x16 cat8(bf16x8 a, bf16x8 b) {
  return __builtin_shufflevector(a, b, 0,1,2,3,4,5,6,7,8,9,10,11,12,13,14,15);
}
static __device__ inline f32x8 wmma_bf16(bf16x16 a, bf16x16 b, f32x8 c) {
  // D = A(16x32 bf16) * B(32x16 bf16) + C(16x16 f32)
  return __builtin_amdgcn_wmma_f32_16x16x32_bf16(false, a, false, b, (short)0, c,
                                                 false, false);
}
static __device__ inline f32x8 zero8() {
  f32x8 z = {0.f,0.f,0.f,0.f,0.f,0.f,0.f,0.f};
  return z;
}

// ---------------------------------------------------------------------------
// Kernel 1: q/k/v = value @ W^T + b  (value = [metal_x ; x] per graph)
// grid (2, 1024, 3), block 128.  Wave = one 16x64 output strip.
// ---------------------------------------------------------------------------
__global__ __launch_bounds__(128) void qkv_proj_kernel(
    const float* __restrict__ x, const float* __restrict__ metal_x,
    const float* __restrict__ Wq, const float* __restrict__ Wk,
    const float* __restrict__ Wv, const float* __restrict__ bq,
    const float* __restrict__ bk, const float* __restrict__ bv,
    __bf16* __restrict__ qws, __bf16* __restrict__ kws, __bf16* __restrict__ vws)
{
  const float* W; const float* bias; __bf16* dst;
  if (blockIdx.z == 0)      { W = Wq; bias = bq; dst = qws; }
  else if (blockIdx.z == 1) { W = Wk; bias = bk; dst = kws; }
  else                      { W = Wv; bias = bv; dst = vws; }

  const int lane = threadIdx.x & 31;
  const int wave = threadIdx.x >> 5;
  const int half = lane >> 4;       // K-half selector (ISA 16-bit A/B layouts)
  const int c    = lane & 15;       // M (A rows) / N (B cols, D cols)
  const int mbase  = blockIdx.y * 16;
  const int nstrip = (blockIdx.x * 4 + wave) * 64;

  // virtual "value" row for this lane (constant across the K loop)
  const int arowi = mbase + c;
  const int gb = arowi >> 8;
  const int gl = arowi & 255;
  const float* __restrict__ arow =
      (gl == 0) ? (metal_x + (size_t)gb * E_DIM)
                : (x + ((size_t)gb * NNODE + (gl - 1)) * E_DIM);

  const float* wrow[4];
  float bias_c[4];
  #pragma unroll
  for (int t = 0; t < 4; ++t) {
    const int j = nstrip + t * 16 + c;
    wrow[t] = W + (size_t)j * E_DIM;
    bias_c[t] = bias[j];
  }

  f32x8 acc[4] = {zero8(), zero8(), zero8(), zero8()};
  for (int kk = 0; kk < E_DIM; kk += 32) {
    // A fragment: lane holds row (mbase+c); K(e) = e + half*16 (contiguous 16)
    const int ka = kk + half * 16;
    f32x8 a0 = *(const f32x8*)(arow + ka);
    f32x8 a1 = *(const f32x8*)(arow + ka + 8);
    bf16x16 af = cat8(__builtin_convertvector(a0, bf16x8),
                      __builtin_convertvector(a1, bf16x8));
    // B fragment: lane holds col; K(e) = e + half*16 (contiguous 16)
    const int kb = kk + half * 16;
    #pragma unroll
    for (int t = 0; t < 4; ++t) {
      f32x8 w0 = *(const f32x8*)(wrow[t] + kb);
      f32x8 w1 = *(const f32x8*)(wrow[t] + kb + 8);
      acc[t] = wmma_bf16(af,
                         cat8(__builtin_convertvector(w0, bf16x8),
                              __builtin_convertvector(w1, bf16x8)),
                         acc[t]);
    }
  }

  // D tile: VGPR r -> row (r + half*8), col = c (per ISA C/D layout)
  #pragma unroll
  for (int t = 0; t < 4; ++t) {
    const int j = nstrip + t * 16 + c;
    #pragma unroll
    for (int r = 0; r < 8; ++r) {
      const int grow = mbase + r + half * 8;
      dst[(size_t)grow * E_DIM + j] = (__bf16)(acc[t][r] + bias_c[t]);
    }
  }
}

// ---------------------------------------------------------------------------
// Kernel 2: attention per (b,h); block = 2 waves x 16 query rows.
// grid (8, H, B), block 64.  V^T staged in LDS; P transposed through LDS.
// ---------------------------------------------------------------------------
__global__ __launch_bounds__(64) void attn_kernel(
    const __bf16* __restrict__ qws, const __bf16* __restrict__ kws,
    const __bf16* __restrict__ vws, __bf16* __restrict__ ows)
{
  __shared__ __align__(16) __bf16 VT[DH][LP];        // V^T: [d][key]
  __shared__ __align__(16) __bf16 P[2][16][LP];      // probs per wave: [q][key]

  const int b = blockIdx.z;
  const int h = blockIdx.y;
  const int lane = threadIdx.x & 31;
  const int wave = threadIdx.x >> 5;
  const int half = lane >> 4;
  const int c    = lane & 15;
  const int mtile = blockIdx.x * 2 + wave;

  // cooperative transpose-stage of the V head slice into LDS
  const __bf16* __restrict__ vbase = vws + (size_t)b * LSEQ * E_DIM + h * DH;
  for (int i = threadIdx.x; i < LSEQ * (DH / 8); i += 64) {
    const int key = i >> 3;
    const int d0  = (i & 7) * 8;
    bf16x8 v = *(const bf16x8*)(vbase + (size_t)key * E_DIM + d0);
    #pragma unroll
    for (int j = 0; j < 8; ++j) VT[d0 + j][key] = v[j];
  }
  __syncthreads();

  // Q: two 16x32 A-fragments covering Dh=64
  const int qrow = mtile * 16 + c;
  const __bf16* __restrict__ qbase =
      qws + ((size_t)b * LSEQ + qrow) * E_DIM + h * DH;
  bf16x16 qf[2];
  #pragma unroll
  for (int kk = 0; kk < 2; ++kk) {
    const int ka = kk * 32 + half * 16;
    qf[kk] = cat8(*(const bf16x8*)(qbase + ka),
                  *(const bf16x8*)(qbase + ka + 8));
  }

  // S = Q K^T : 16 C-tiles of 16 keys, f32 accumulate
  f32x8 sc[16];
  #pragma unroll
  for (int n = 0; n < 16; ++n) {
    const int key = n * 16 + c;
    const __bf16* __restrict__ krow =
        kws + ((size_t)b * LSEQ + key) * E_DIM + h * DH;
    f32x8 a = zero8();
    #pragma unroll
    for (int kk = 0; kk < 2; ++kk) {
      const int kd = kk * 32 + half * 16;
      a = wmma_bf16(qf[kk],
                    cat8(*(const bf16x8*)(krow + kd),
                         *(const bf16x8*)(krow + kd + 8)),
                    a);
    }
    sc[n] = a;
  }

  // f32 softmax per row; row spans 16 lanes of one half-wave + 16 tiles
  const float scale = 0.125f;  // 1/sqrt(64)
  #pragma unroll
  for (int r = 0; r < 8; ++r) {
    float mx = -3.0e38f;
    #pragma unroll
    for (int n = 0; n < 16; ++n) { sc[n][r] *= scale; mx = fmaxf(mx, sc[n][r]); }
    #pragma unroll
    for (int msk = 1; msk < 16; msk <<= 1) mx = fmaxf(mx, __shfl_xor(mx, msk, 32));
    float sum = 0.f;
    #pragma unroll
    for (int n = 0; n < 16; ++n) { float p = __expf(sc[n][r] - mx); sc[n][r] = p; sum += p; }
    #pragma unroll
    for (int msk = 1; msk < 16; msk <<= 1) sum += __shfl_xor(sum, msk, 32);
    const float inv = 1.f / sum;
    const int qm = r + half * 8;
    #pragma unroll
    for (int n = 0; n < 16; ++n)
      P[wave][qm][n * 16 + c] = (__bf16)(sc[n][r] * inv);
  }
  // per-wave LDS region; DS ops in-order within a wave -> no barrier needed

  // O = P V : 4 d-tiles, K over 256 keys in 8 groups of 32
  f32x8 oacc[4] = {zero8(), zero8(), zero8(), zero8()};
  for (int g = 0; g < 8; ++g) {
    const int ka = g * 32 + half * 16;
    bf16x16 af = cat8(*(const bf16x8*)(&P[wave][c][ka]),
                      *(const bf16x8*)(&P[wave][c][ka + 8]));
    const int kb2 = g * 32 + half * 16;
    #pragma unroll
    for (int t = 0; t < 4; ++t) {
      const int d = t * 16 + c;
      oacc[t] = wmma_bf16(af,
                          cat8(*(const bf16x8*)(&VT[d][kb2]),
                               *(const bf16x8*)(&VT[d][kb2 + 8])),
                          oacc[t]);
    }
  }

  __bf16* __restrict__ obase =
      ows + ((size_t)b * LSEQ + mtile * 16) * E_DIM + h * DH;
  #pragma unroll
  for (int t = 0; t < 4; ++t)
    #pragma unroll
    for (int r = 0; r < 8; ++r)
      obase[(size_t)(r + half * 8) * E_DIM + t * 16 + c] = (__bf16)oacc[t][r];
}

// ---------------------------------------------------------------------------
// Kernel 3: out = O @ Wo^T + bo, scattered to the reference [L,B,E] ordering.
// grid (2, 1024, 1), block 128.
// ---------------------------------------------------------------------------
__global__ __launch_bounds__(128) void out_proj_kernel(
    const __bf16* __restrict__ ows, const float* __restrict__ Wo,
    const float* __restrict__ bo, float* __restrict__ out)
{
  const int lane = threadIdx.x & 31;
  const int wave = threadIdx.x >> 5;
  const int half = lane >> 4;
  const int c    = lane & 15;
  const int mbase  = blockIdx.y * 16;
  const int nstrip = (blockIdx.x * 4 + wave) * 64;

  const __bf16* __restrict__ arow = ows + (size_t)(mbase + c) * E_DIM;
  const float* wrow[4];
  float bias_c[4];
  #pragma unroll
  for (int t = 0; t < 4; ++t) {
    const int j = nstrip + t * 16 + c;
    wrow[t] = Wo + (size_t)j * E_DIM;
    bias_c[t] = bo[j];
  }

  f32x8 acc[4] = {zero8(), zero8(), zero8(), zero8()};
  for (int kk = 0; kk < E_DIM; kk += 32) {
    const int ka = kk + half * 16;
    bf16x16 af = cat8(*(const bf16x8*)(arow + ka),
                      *(const bf16x8*)(arow + ka + 8));
    const int kb = kk + half * 16;
    #pragma unroll
    for (int t = 0; t < 4; ++t) {
      f32x8 w0 = *(const f32x8*)(wrow[t] + kb);
      f32x8 w1 = *(const f32x8*)(wrow[t] + kb + 8);
      acc[t] = wmma_bf16(af,
                         cat8(__builtin_convertvector(w0, bf16x8),
                              __builtin_convertvector(w1, bf16x8)),
                         acc[t]);
    }
  }

  // scatter: row (b,l) -> l==0 ? metal block (tail) : ((l-1)*B + b)
  #pragma unroll
  for (int r = 0; r < 8; ++r) {
    const int grow = mbase + r + half * 8;
    const int gb = grow >> 8;
    const int gl = grow & 255;
    float* __restrict__ drow =
        (gl == 0) ? (out + ((size_t)NB * NNODE + gb) * E_DIM)
                  : (out + (((size_t)(gl - 1)) * NB + gb) * E_DIM);
    #pragma unroll
    for (int t = 0; t < 4; ++t)
      drow[nstrip + t * 16 + c] = acc[t][r] + bias_c[t];
  }
}

// ---------------------------------------------------------------------------
extern "C" void kernel_launch(void* const* d_in, const int* in_sizes, int n_in,
                              void* d_out, int out_size, void* d_ws, size_t ws_size,
                              hipStream_t stream) {
  (void)in_sizes; (void)n_in; (void)out_size; (void)ws_size;
  const float* x       = (const float*)d_in[0];
  const float* metal_x = (const float*)d_in[1];
  // d_in[2] = batch indices (structure is implicit; unused)
  const float* Wq = (const float*)d_in[3];
  const float* Wk = (const float*)d_in[4];
  const float* Wv = (const float*)d_in[5];
  const float* bq = (const float*)d_in[6];
  const float* bk = (const float*)d_in[7];
  const float* bv = (const float*)d_in[8];
  const float* Wo = (const float*)d_in[9];
  const float* bo = (const float*)d_in[10];
  float* out = (float*)d_out;

  const size_t per = (size_t)NB * LSEQ * E_DIM;   // 8.4M elems per tensor
  __bf16* qws = (__bf16*)d_ws;                    // 4 x 16 MB bf16 workspace
  __bf16* kws = qws + per;
  __bf16* vws = kws + per;
  __bf16* ows = vws + per;

  qkv_proj_kernel<<<dim3(2, (NB * LSEQ) / 16, 3), 128, 0, stream>>>(
      x, metal_x, Wq, Wk, Wv, bq, bk, bv, qws, kws, vws);
  attn_kernel<<<dim3(LSEQ / 32, NHEAD, NB), 64, 0, stream>>>(qws, kws, vws, ows);
  out_proj_kernel<<<dim3(2, (NB * LSEQ) / 16, 1), 128, 0, stream>>>(ows, Wo, bo, out);
}
